// RecursiveNN_5617817223307
// MI455X (gfx1250) — matrix operand order repaired and checked by reading, not compiled, not run on hardware
//
#include <hip/hip_runtime.h>

// ---------------------------------------------------------------------------
// TreeLSTM (DEPTH=10, B=64, D=512) for gfx1250 using bf16 WMMA.
//   x = [h_left | h_right] (K=1024) ; iou = x@W_iou+b_iou ; f = sig(x@W_f+b_f)
//   c' = sig(i)*tanh(u) + f_l*c_l + f_r*c_r ; h' = sig(o)*tanh(c')
// One wave computes a 16(row)x16(col j) tile: 5 fused WMMA chains (i,o,u,fl,fr)
// sharing the A fragment, K=1024 -> 160 v_wmma_f32_16x16x32_bf16 per wave.
// ---------------------------------------------------------------------------

typedef __attribute__((ext_vector_type(16))) __bf16 v16bf;
typedef __attribute__((ext_vector_type(8)))  __bf16 bf16x8;
typedef __attribute__((ext_vector_type(8)))  float  v8f;

#define D_MODEL   512
#define BATCH     64
#define N_LEAVES  1024

// workspace layout (bytes)
#define OFF_HA    0ull                               // bf16 [1024][64][512]  64 MiB
#define OFF_HB    67108864ull                        // bf16 [ 512][64][512]  32 MiB
#define OFF_CA    100663296ull                       // f32  [ 256][64][512]  32 MiB
#define OFF_CB    134217728ull                       // f32  [ 512][64][512]  64 MiB
#define OFF_WIOUT 201326592ull                       // bf16 [1536][1024]      3 MiB
#define OFF_WFT   204472320ull                       // bf16 [1024][1024]      2 MiB

__device__ __forceinline__ float sigmf(float x) {
    return 1.0f / (1.0f + __expf(-x));
}

// A fragment: lane<16 -> row M=lane, K = base+0..15 (elems 0..15)
//             lane>=16 -> same row, K base shifted by +16 (applied by caller).
__device__ __forceinline__ v16bf load_frag_a(const __bf16* p) {
    bf16x8 lo = *(const bf16x8*)(p);
    bf16x8 hi = *(const bf16x8*)(p + 8);
    v16bf r;
#pragma unroll
    for (int e = 0; e < 8; ++e) { r[e] = lo[e]; r[e + 8] = hi[e]; }
    return r;
}

// B fragment (W pre-transposed, column-major in K): 16 contiguous bf16 per lane.
__device__ __forceinline__ v16bf load_frag_b(const __bf16* p) {
    bf16x8 lo = *(const bf16x8*)(p);
    bf16x8 hi = *(const bf16x8*)(p + 8);
    v16bf r;
#pragma unroll
    for (int e = 0; e < 8; ++e) { r[e] = lo[e]; r[e + 8] = hi[e]; }
    return r;
}

__device__ __forceinline__ v8f wmma_bf16(v16bf a, v16bf b, v8f c) {
    return __builtin_amdgcn_wmma_f32_16x16x32_bf16(
        /*neg_a=*/false, a, /*neg_b=*/false, b,
        /*c_mod=*/(short)0, c, /*reuse_a=*/false, /*reuse_b=*/false);
}

// ---- weight convert + transpose: W[k][n] f32 -> WT[n][k] bf16 ---------------
__global__ void tl_convert_weights(const float* __restrict__ W_iou,
                                   const float* __restrict__ W_f,
                                   __bf16* __restrict__ WiouT,
                                   __bf16* __restrict__ WfT) {
    const unsigned NIOU = 1536u * 1024u;
    const unsigned NF   = 1024u * 1024u;
    unsigned idx = blockIdx.x * 256u + threadIdx.x;
    if (idx < NIOU) {
        unsigned n = idx >> 10, k = idx & 1023u;
        WiouT[idx] = (__bf16)W_iou[(size_t)k * 1536u + n];
    } else if (idx < NIOU + NF) {
        unsigned j = idx - NIOU;
        unsigned n = j >> 10, k = j & 1023u;
        WfT[j] = (__bf16)W_f[(size_t)k * 1024u + n];
    }
}

// ---- leaf embedding gather: hLeaf[leaf][b][d] = bf16(emb[tok[b][leaf]][d]) --
__global__ void tl_embed_leaves(const int* __restrict__ tokens,
                                const float* __restrict__ emb,
                                __bf16* __restrict__ hLeaf) {
    int b  = blockIdx.x >> 10;
    int li = blockIdx.x & 1023;
    int tok = tokens[b * N_LEAVES + li];
    const float* e = emb + (size_t)tok * D_MODEL;
    __bf16* dst = hLeaf + ((size_t)li * BATCH + b) * D_MODEL;
#pragma unroll
    for (int d = threadIdx.x; d < D_MODEL; d += 256)
        dst[d] = (__bf16)e[d];
}

// ---- one tree level: grid = 16*n blocks x 256 thr (8 waves, 1 tile/wave) ---
__global__ void __launch_bounds__(256)
tl_level_kernel(const __bf16* __restrict__ hChild,  // [2n][64][512] bf16
                const float*  __restrict__ cChild,  // [2n][64][512] f32 (unused if leaf)
                __bf16* __restrict__ hParent,       // [n][64][512]
                float*  __restrict__ cParent,       // [n][64][512]
                const __bf16* __restrict__ WiouT,   // [1536][1024]
                const __bf16* __restrict__ WfT,     // [1024][1024]
                const float* __restrict__ b_iou,
                const float* __restrict__ b_f,
                int leafLevel) {
    const int wid   = threadIdx.x >> 5;
    const int lane  = threadIdx.x & 31;
    const int tile  = blockIdx.x * 8 + wid;     // 128*n tiles total, exact
    const int mTile = tile >> 5;                // 0 .. 4n-1
    const int jTile = tile & 31;                // 0 .. 31
    const int p     = mTile >> 2;               // parent index in level
    const int b0    = (mTile & 3) << 4;         // batch base of 16-row tile
    const int half  = lane >> 4;
    const int mrow  = lane & 15;
    const int j     = jTile * 16 + mrow;        // output feature column

    // A-fragment base pointers (left / right child rows for this lane)
    const int brow = b0 + mrow;
    const __bf16* aL = hChild + ((size_t)(2 * p)     * BATCH + brow) * D_MODEL + half * 16;
    const __bf16* aR = hChild + ((size_t)(2 * p + 1) * BATCH + brow) * D_MODEL + half * 16;

    // B-fragment base pointers (pre-transposed weights)
    const size_t koff = (size_t)(half * 16);
    const __bf16* bI  = WiouT + (size_t)(j)        * 1024 + koff;
    const __bf16* bO  = WiouT + (size_t)(j + 512)  * 1024 + koff;
    const __bf16* bU  = WiouT + (size_t)(j + 1024) * 1024 + koff;
    const __bf16* bFl = WfT   + (size_t)(j)        * 1024 + koff;
    const __bf16* bFr = WfT   + (size_t)(j + 512)  * 1024 + koff;

    v8f accI = {}, accO = {}, accU = {}, accFl = {}, accFr = {};

#pragma unroll
    for (int s = 0; s < 2; ++s) {                 // s=0: left child K[0,512), s=1: right
        const __bf16* aRow = s ? aR : aL;
        const size_t  kw   = (size_t)s * 512;
#pragma unroll 4
        for (int kk = 0; kk < 512; kk += 32) {
            v16bf A = load_frag_a(aRow + kk);
            v16bf Bi  = load_frag_b(bI  + kw + kk);
            v16bf Bo  = load_frag_b(bO  + kw + kk);
            v16bf Bu  = load_frag_b(bU  + kw + kk);
            v16bf Bfl = load_frag_b(bFl + kw + kk);
            v16bf Bfr = load_frag_b(bFr + kw + kk);
            accI  = wmma_bf16(A, Bi,  accI);
            accO  = wmma_bf16(A, Bo,  accO);
            accU  = wmma_bf16(A, Bu,  accU);
            accFl = wmma_bf16(A, Bfl, accFl);
            accFr = wmma_bf16(A, Bfr, accFr);
        }
    }

    // fused LSTM gates + parent write-out
    const float biasI  = b_iou[j];
    const float biasO  = b_iou[j + 512];
    const float biasU  = b_iou[j + 1024];
    const float biasFl = b_f[j];
    const float biasFr = b_f[j + 512];

    const float* cL = cChild + ((size_t)(2 * p)     * BATCH) * D_MODEL;
    const float* cR = cChild + ((size_t)(2 * p + 1) * BATCH) * D_MODEL;
    float*  cP = cParent + ((size_t)p * BATCH) * D_MODEL;
    __bf16* hP = hParent + ((size_t)p * BATCH) * D_MODEL;

#pragma unroll
    for (int v = 0; v < 8; ++v) {
        const int b = b0 + v + half * 8;          // C/D layout: VGPR v, lane-half
        float ig = sigmf(accI[v]  + biasI);
        float og = sigmf(accO[v]  + biasO);
        float ug = tanhf(accU[v]  + biasU);
        float fl = sigmf(accFl[v] + biasFl);
        float fr = sigmf(accFr[v] + biasFr);
        float cn = ig * ug;
        if (!leafLevel) {
            cn += fl * cL[(size_t)b * D_MODEL + j] + fr * cR[(size_t)b * D_MODEL + j];
        }
        float hn = og * tanhf(cn);
        cP[(size_t)b * D_MODEL + j] = cn;
        hP[(size_t)b * D_MODEL + j] = (__bf16)hn;
    }
}

// ---- final projection: out[b][cls] = h_root[b] . W_out[:,cls] --------------
__global__ void tl_out_proj(const __bf16* __restrict__ hRoot,
                            const float* __restrict__ W_out,
                            float* __restrict__ out) {
    int t = threadIdx.x;
    if (t < BATCH * 2) {
        int b = t >> 1, cls = t & 1;
        const __bf16* h = hRoot + (size_t)b * D_MODEL;
        float s = 0.0f;
        for (int d = 0; d < D_MODEL; ++d)
            s += (float)h[d] * W_out[d * 2 + cls];
        out[t] = s;
    }
}

extern "C" void kernel_launch(void* const* d_in, const int* in_sizes, int n_in,
                              void* d_out, int out_size, void* d_ws, size_t ws_size,
                              hipStream_t stream) {
    const int*   tokens    = (const int*)d_in[0];
    const float* embedding = (const float*)d_in[1];
    const float* W_iou     = (const float*)d_in[2];
    const float* b_iou     = (const float*)d_in[3];
    const float* W_f       = (const float*)d_in[4];
    const float* b_f       = (const float*)d_in[5];
    const float* W_out     = (const float*)d_in[6];

    char* ws = (char*)d_ws;
    __bf16* hA    = (__bf16*)(ws + OFF_HA);
    __bf16* hB    = (__bf16*)(ws + OFF_HB);
    float*  cA    = (float*)(ws + OFF_CA);
    float*  cB    = (float*)(ws + OFF_CB);
    __bf16* WiouT = (__bf16*)(ws + OFF_WIOUT);
    __bf16* WfT   = (__bf16*)(ws + OFF_WFT);

    // 1) weights -> bf16, transposed (K-contiguous per output column)
    tl_convert_weights<<<10240, 256, 0, stream>>>(W_iou, W_f, WiouT, WfT);

    // 2) leaves: h = bf16(embedding[tokens]); c == 0 handled by leaf flag
    tl_embed_leaves<<<BATCH * N_LEAVES, 256, 0, stream>>>(tokens, embedding, hA);

    // 3) levels 9..0, ping-pong A/B buffers (children in A when level is odd)
    for (int level = 9; level >= 0; --level) {
        int n = 1 << level;
        bool childInA = (level & 1) != 0;
        const __bf16* hC = childInA ? hA : hB;
        const float*  cC = childInA ? cA : cB;
        __bf16* hP = childInA ? hB : hA;
        float*  cP = childInA ? cB : cA;
        tl_level_kernel<<<16 * n, 256, 0, stream>>>(
            hC, cC, hP, cP, WiouT, WfT, b_iou, b_f, (level == 9) ? 1 : 0);
    }

    // 4) root projection (root h landed in hA, node 0)
    tl_out_proj<<<1, 128, 0, stream>>>(hA, W_out, (float*)d_out);
}
